// MultiHeadAttention_37967510896772
// MI455X (gfx1250) — matrix operand order repaired and checked by reading, not compiled, not run on hardware
//
#include <hip/hip_runtime.h>
#include <hip/hip_fp16.h>

typedef _Float16 half_t;
typedef __attribute__((ext_vector_type(16))) _Float16 v16h;
typedef __attribute__((ext_vector_type(8)))  float    v8f;

#define BB 2
#define TT 4096
#define DD 512
#define HH 8
#define HD 64

// ---------------------------------------------------------------------------
// gfx1250 async copy helpers: per-lane 16B global->LDS via the async engine
// (ASYNCcnt), waited with s_wait_asynccnt.  Low 32 bits of a generic pointer
// to __shared__ are the LDS byte offset (shared aperture lives in the high
// 32 bits on CDNA5).  ASYNCcnt completes in order, so waiting for <= N leaves
// the newest N copies in flight while guaranteeing older tiles have landed.
// ---------------------------------------------------------------------------
__device__ __forceinline__ void async_copy16(const void* g, void* l) {
    unsigned            loff = (unsigned)(size_t)l;
    unsigned long long  ga   = (unsigned long long)(size_t)g;
    asm volatile("global_load_async_to_lds_b128 %0, %1, off"
                 :: "v"(loff), "v"(ga) : "memory");
}
__device__ __forceinline__ void wait_async0() {
    asm volatile("s_wait_asynccnt 0" ::: "memory");
}
__device__ __forceinline__ void wait_async1() {
    asm volatile("s_wait_asynccnt 1" ::: "memory");
}
__device__ __forceinline__ void wait_async2() {
    asm volatile("s_wait_asynccnt 2" ::: "memory");
}

// ---------------------------------------------------------------------------
// Kernel 1: projections.  mode = blockIdx.z: 0->Q (scaled 1/8), 1->K, 2->V^T
// out[m][n] = sum_k X[m][k] * W[n][k]   (x @ W.T)
// ---------------------------------------------------------------------------
__global__ __launch_bounds__(256)
void proj_kernel(const float* __restrict__ Xq, const float* __restrict__ Xk,
                 const float* __restrict__ Xv,
                 const float* __restrict__ Wq, const float* __restrict__ Wk,
                 const float* __restrict__ Wv,
                 half_t* __restrict__ Qf, half_t* __restrict__ Kf,
                 half_t* __restrict__ Vt)
{
    const int mode = blockIdx.z;
    const float* X = (mode == 0) ? Xq : (mode == 1 ? Xk : Xv);
    const float* W = (mode == 0) ? Wq : (mode == 1 ? Wk : Wv);

    const int lane = threadIdx.x & 31;
    const int wave = threadIdx.x >> 5;
    const int m0 = blockIdx.y * 16;               // 16 rows of X per block
    const int n0 = blockIdx.x * 128 + wave * 16;  // 16 output cols per wave

    // A-fragment (16x32 f16): lanes 0-15 rows M, K=0..15;
    // lanes 16-31 same rows, K=16..31
    const int arow  = m0 + (lane & 15);
    const int abase = (lane >> 4) * 16;
    // B-fragment (32x16): lanes 0-15 col N K=0..15; lanes 16-31 col N K=16..31
    const int bcol  = n0 + (lane & 15);
    const int bbase = (lane >> 4) * 16;

    v8f c = {};
    for (int k = 0; k < DD; k += 32) {
        v16h a;
        {
            const float* p0 = X + (size_t)arow * DD + k + abase;
            #pragma unroll
            for (int j = 0; j < 8; ++j) {
                a[j]     = (half_t)p0[j];
                a[j + 8] = (half_t)p0[8 + j];
            }
        }
        v16h b;
        {
            const float* p = W + (size_t)bcol * DD + k + bbase;
            #pragma unroll
            for (int j = 0; j < 16; ++j) b[j] = (half_t)p[j];
        }
        c = __builtin_amdgcn_wmma_f32_16x16x32_f16(false, a, false, b,
                                                   (short)0, c, false, false);
    }

    // C/D layout: lane holds col = lane&15, rows r + 8*(lane>=16)
    const int ccol  = n0 + (lane & 15);
    const int rbase = (lane >> 4) * 8;
    if (mode < 2) {
        half_t* Obuf = (mode == 0) ? Qf : Kf;
        const float scale = (mode == 0) ? 0.125f : 1.0f;  // 1/sqrt(64) on Q
        #pragma unroll
        for (int r = 0; r < 8; ++r) {
            const int row = m0 + rbase + r;
            Obuf[(size_t)row * DD + ccol] = (half_t)(c[r] * scale);
        }
    } else {
        // V transposed per head: Vt[b][h][hd][T]
        const int h  = ccol >> 6;
        const int dd = ccol & 63;
        #pragma unroll
        for (int r = 0; r < 8; ++r) {
            const int row = m0 + rbase + r;        // global token over B*T
            const int bb = row >> 12;
            const int tt = row & (TT - 1);
            Vt[(((size_t)bb * HH + h) * HD + dd) * TT + tt] = (half_t)c[r];
        }
    }
}

// ---------------------------------------------------------------------------
// Kernel 2: flash-style attention. One wave owns a 16-row query tile; all 8
// waves of a block share one (b,h) slice, so K/V tiles are staged once per
// block into double-buffered LDS via the gfx1250 async-to-LDS engine
// (prefetch tile i+1 while WMMA consumes tile i).
// Pass 1: online max/sum over all keys.  Pass 2: recompute S, write normalized
// probs (non-temporal) to d_out, accumulate Z = P @ V via WMMA.
// ---------------------------------------------------------------------------
__global__ __launch_bounds__(256)
void attn_kernel(const half_t* __restrict__ Qf, const half_t* __restrict__ Kf,
                 const half_t* __restrict__ Vt, half_t* __restrict__ Zf,
                 float* __restrict__ attn_out)
{
    __shared__ half_t ldsK[2][32 * 64];   // ping-pong 32 keys x 64 dims (8KB)
    __shared__ half_t ldsV[2][64 * 32];   // ping-pong 64 dims x 32 keys (8KB)
    __shared__ half_t pbuf[8][16 * 32];   // wave-private p staging     (8KB)

    const int tid  = threadIdx.x;
    const int lane = tid & 31;
    const int wave = tid >> 5;
    const int widx = blockIdx.x * 8 + wave;      // global q-tile id
    const int b  = widx >> 11;
    const int h  = (widx >> 8) & 7;
    const int q0 = (widx & 255) * 16;

    const half_t* Qbase = Qf + (size_t)b * TT * DD + h * HD;
    const half_t* Kbase = Kf + (size_t)b * TT * DD + h * HD;
    const half_t* Vbase = Vt + ((size_t)b * HH + h) * HD * TT;

    // cooperative staging assignments (16B per thread per tile)
    const int krow = tid >> 3;            // 0..31 : key row within K tile
    const int kchk = (tid & 7) * 8;       // dim chunk (8 halves)
    const int vrow = tid >> 2;            // 0..63 : dim row within V tile
    const int vchk = (tid & 3) * 8;       // key chunk (8 halves)
    const int koff = krow * 64 + kchk;
    const int voff = vrow * 32 + vchk;

    const int abase = (lane >> 4) * 16;
    const int arow  = q0 + (lane & 15);
    const int brow  = lane & 15;
    const int bbase = (lane >> 4) * 16;
    const int rbase = (lane >> 4) * 8;

    // Persistent Q A-fragments (head_dim 64 -> two 16x32 fragments)
    v16h qa0, qa1;
    {
        const half_t* p = Qbase + (size_t)arow * DD;
        #pragma unroll
        for (int j = 0; j < 8; ++j) {
            qa0[j]     = p[abase + j];
            qa0[j + 8] = p[abase + 8 + j];
            qa1[j]     = p[32 + abase + j];
            qa1[j + 8] = p[32 + abase + 8 + j];
        }
    }

    // ---- Pass 1: per-lane online softmax statistics ----
    float m[8], s[8];
    #pragma unroll
    for (int r = 0; r < 8; ++r) { m[r] = -3.0e38f; s[r] = 0.0f; }

    async_copy16(Kbase + (size_t)krow * DD + kchk, &ldsK[0][koff]);
    int buf = 0;
    for (int s0 = 0; s0 < TT; s0 += 32, buf ^= 1) {
        if (s0 + 32 < TT) {
            async_copy16(Kbase + (size_t)(s0 + 32 + krow) * DD + kchk,
                         &ldsK[buf ^ 1][koff]);
            wait_async1();           // oldest (current tile) has landed
        } else {
            wait_async0();
        }
        __syncthreads();
        #pragma unroll
        for (int t = 0; t < 2; ++t) {
            const half_t* p = &ldsK[buf][(t * 16 + brow) * 64 + bbase];
            v16h kb0, kb1;
            #pragma unroll
            for (int j = 0; j < 16; ++j) { kb0[j] = p[j]; kb1[j] = p[32 + j]; }
            v8f c = {};
            c = __builtin_amdgcn_wmma_f32_16x16x32_f16(false, qa0, false, kb0,
                                                       (short)0, c, false, false);
            c = __builtin_amdgcn_wmma_f32_16x16x32_f16(false, qa1, false, kb1,
                                                       (short)0, c, false, false);
            #pragma unroll
            for (int r = 0; r < 8; ++r) {
                const float x  = c[r];
                const float mn = fmaxf(m[r], x);
                s[r] = s[r] * __expf(m[r] - mn) + __expf(x - mn);
                m[r] = mn;
            }
        }
        __syncthreads();   // reads done before buffer is overwritten again
    }
    // combine across the 16 lanes of each half-group (cols of each row)
    #pragma unroll
    for (int off = 1; off < 16; off <<= 1) {
        #pragma unroll
        for (int r = 0; r < 8; ++r) {
            const float mo = __shfl_xor(m[r], off, 32);
            const float so = __shfl_xor(s[r], off, 32);
            const float mn = fmaxf(m[r], mo);
            s[r] = s[r] * __expf(m[r] - mn) + so * __expf(mo - mn);
            m[r] = mn;
        }
    }
    float rinv[8];
    #pragma unroll
    for (int r = 0; r < 8; ++r) rinv[r] = 1.0f / s[r];

    // ---- Pass 2: recompute, normalize, emit attn, accumulate Z = P @ V ----
    v8f z[4] = {v8f{}, v8f{}, v8f{}, v8f{}};
    half_t* myp = &pbuf[wave][0];
    float* attn_row0 = attn_out + ((size_t)(b * HH + h) * TT + q0) * TT;

    async_copy16(Kbase + (size_t)krow * DD + kchk, &ldsK[0][koff]);
    async_copy16(Vbase + (size_t)vrow * TT + vchk, &ldsV[0][voff]);
    buf = 0;
    for (int s0 = 0; s0 < TT; s0 += 32, buf ^= 1) {
        if (s0 + 32 < TT) {
            async_copy16(Kbase + (size_t)(s0 + 32 + krow) * DD + kchk,
                         &ldsK[buf ^ 1][koff]);
            async_copy16(Vbase + (size_t)vrow * TT + s0 + 32 + vchk,
                         &ldsV[buf ^ 1][voff]);
            wait_async2();           // current K+V pair has landed
        } else {
            wait_async0();
        }
        __syncthreads();

        #pragma unroll
        for (int t = 0; t < 2; ++t) {
            const half_t* p = &ldsK[buf][(t * 16 + brow) * 64 + bbase];
            v16h kb0, kb1;
            #pragma unroll
            for (int j = 0; j < 16; ++j) { kb0[j] = p[j]; kb1[j] = p[32 + j]; }
            v8f c = {};
            c = __builtin_amdgcn_wmma_f32_16x16x32_f16(false, qa0, false, kb0,
                                                       (short)0, c, false, false);
            c = __builtin_amdgcn_wmma_f32_16x16x32_f16(false, qa1, false, kb1,
                                                       (short)0, c, false, false);
            const int col = t * 16 + (lane & 15);
            #pragma unroll
            for (int r = 0; r < 8; ++r) {
                const float pv = __expf(c[r] - m[r]) * rinv[r];
                const int row = rbase + r;
                // streaming 1GB output: non-temporal store
                __builtin_nontemporal_store(
                    pv, &attn_row0[(size_t)row * TT + s0 + col]);
                myp[row * 32 + col] = (half_t)pv;
            }
        }
        // wave-local LDS RAW on pbuf: LDS ops are in-order per wave
        asm volatile("s_wait_dscnt 0" ::: "memory");

        // A-fragment = p tile (16x32 f16) from LDS
        v16h pa;
        {
            const half_t* pp = myp + (lane & 15) * 32 + abase;
            #pragma unroll
            for (int j = 0; j < 8; ++j) { pa[j] = pp[j]; pa[j + 8] = pp[8 + j]; }
        }
        // 4 B-fragments from the staged V tile, accumulate Z
        #pragma unroll
        for (int n = 0; n < 4; ++n) {
            const half_t* vp = &ldsV[buf][(n * 16 + (lane & 15)) * 32 + bbase];
            v16h vb;
            #pragma unroll
            for (int j = 0; j < 16; ++j) vb[j] = vp[j];
            z[n] = __builtin_amdgcn_wmma_f32_16x16x32_f16(
                false, pa, false, vb, (short)0, z[n], false, false);
        }
        __syncthreads();   // LDS reads done before next async overwrite
    }

    // Z tile -> f16 workspace [B*T, D], head columns h*64..h*64+63
    #pragma unroll
    for (int n = 0; n < 4; ++n) {
        #pragma unroll
        for (int r = 0; r < 8; ++r) {
            const int row = q0 + rbase + r;
            Zf[((size_t)b * TT + row) * DD + h * HD + n * 16 + (lane & 15)] =
                (half_t)z[n][r];
        }
    }
}

// ---------------------------------------------------------------------------
// Kernel 3: out = z @ Wo.T + bo  (fp32 output)
// ---------------------------------------------------------------------------
__global__ __launch_bounds__(256)
void outproj_kernel(const half_t* __restrict__ Zf, const float* __restrict__ Wo,
                    const float* __restrict__ bo, float* __restrict__ Out)
{
    const int lane = threadIdx.x & 31;
    const int wave = threadIdx.x >> 5;
    const int m0 = blockIdx.y * 16;
    const int n0 = blockIdx.x * 128 + wave * 16;

    const int arow  = m0 + (lane & 15);
    const int abase = (lane >> 4) * 16;
    const int bcol  = n0 + (lane & 15);
    const int bbase = (lane >> 4) * 16;

    v8f c = {};
    for (int k = 0; k < DD; k += 32) {
        v16h a;
        {
            const half_t* pa = Zf + (size_t)arow * DD + k + abase;
            #pragma unroll
            for (int j = 0; j < 8; ++j) { a[j] = pa[j]; a[j + 8] = pa[8 + j]; }
        }
        v16h b;
        {
            const float* pb = Wo + (size_t)bcol * DD + k + bbase;
            #pragma unroll
            for (int j = 0; j < 16; ++j) b[j] = (half_t)pb[j];
        }
        c = __builtin_amdgcn_wmma_f32_16x16x32_f16(false, a, false, b,
                                                   (short)0, c, false, false);
    }

    const int ccol  = n0 + (lane & 15);
    const float bias = bo[ccol];
    const int rbase = (lane >> 4) * 8;
    #pragma unroll
    for (int r = 0; r < 8; ++r)
        Out[(size_t)(m0 + rbase + r) * DD + ccol] = c[r] + bias;
}

// ---------------------------------------------------------------------------
extern "C" void kernel_launch(void* const* d_in, const int* in_sizes, int n_in,
                              void* d_out, int out_size, void* d_ws,
                              size_t ws_size, hipStream_t stream)
{
    const float* q  = (const float*)d_in[0];
    const float* k  = (const float*)d_in[1];
    const float* v  = (const float*)d_in[2];
    const float* Wq = (const float*)d_in[3];
    const float* Wk = (const float*)d_in[4];
    const float* Wv = (const float*)d_in[5];
    const float* Wo = (const float*)d_in[6];
    const float* bo = (const float*)d_in[7];

    float* out  = (float*)d_out;                       // [B,T,D]
    float* attn = out + (size_t)BB * TT * DD;          // [B,H,T,T]

    half_t* Qf = (half_t*)d_ws;                        // [B*T, D] f16 (scaled)
    half_t* Kf = Qf + (size_t)BB * TT * DD;            // [B*T, D] f16
    half_t* Vt = Kf + (size_t)BB * TT * DD;            // [B, H, hd, T] f16
    half_t* Zf = Vt + (size_t)BB * TT * DD;            // [B*T, D] f16

    dim3 blk(256);
    proj_kernel<<<dim3(DD / 128, (BB * TT) / 16, 3), blk, 0, stream>>>(
        q, k, v, Wq, Wk, Wv, Qf, Kf, Vt);
    attn_kernel<<<dim3((BB * HH * (TT / 16)) / 8), blk, 0, stream>>>(
        Qf, Kf, Vt, Zf, attn);
    outproj_kernel<<<dim3(DD / 128, (BB * TT) / 16, 1), blk, 0, stream>>>(
        Zf, Wo, bo, out);
}
